// BiBoAttention_47974784697177
// MI455X (gfx1250) — compile-verified, matrix operand order checked by reading, not run on hardware
//
#include <hip/hip_runtime.h>
#include <hip/hip_bf16.h>
#include <cstdint>

// ---------------------------------------------------------------------------
// BiBo attention forward for MI455X (gfx1250), wave32.
// - All contractions via v_wmma_f32_16x16x32_bf16 (f32 accumulate).
// - GEMMs: 128x128 block tiles, LDS-staged via GLOBAL_LOAD_ASYNC_TO_LDS_B128
//   (ASYNCcnt-tracked, double-buffered) -> ~64 FLOP/byte HBM intensity.
// - Attention: flash-style, online softmax, P relayout through LDS.
// ---------------------------------------------------------------------------

#define B_   2
#define S_   2048
#define H_   2048
#define NH_  16
#define NKV_ 4
#define HD_  128
#define NEGV (-1e9f)
#define EPSV (1e-6f)

typedef __attribute__((ext_vector_type(16))) __bf16 bf16x16;
typedef __attribute__((ext_vector_type(8)))  __bf16 bf16x8;
typedef __attribute__((ext_vector_type(8)))  float  f32x8;

__device__ __forceinline__ unsigned short f32_to_bf16(float f) {
  unsigned int u = __float_as_uint(f);
  unsigned int r = u + 0x7FFFu + ((u >> 16) & 1u);   // round-to-nearest-even
  return (unsigned short)(r >> 16);
}

// 16-bit WMMA fragment: per lane, elems 0..7 = contiguous K run (kh*8..),
// elems 8..15 = contiguous K run (16+kh*8..). Two 16B-aligned b128 loads.
__device__ __forceinline__ bf16x16 load_frag(const unsigned short* p0,
                                             const unsigned short* p1) {
  bf16x8 a = *(const bf16x8*)p0;
  bf16x8 b = *(const bf16x8*)p1;
  bf16x16 r;
#pragma unroll
  for (int i = 0; i < 8; ++i) { r[i] = a[i]; r[i + 8] = b[i]; }
  return r;
}

__device__ __forceinline__ f32x8 wmma_bf16(bf16x16 a, bf16x16 b, f32x8 c) {
  return __builtin_amdgcn_wmma_f32_16x16x32_bf16(
      /*neg_a=*/false, a, /*neg_b=*/false, b,
      /*c_mod=*/(short)0, c, /*reuse_a=*/false, /*reuse_b=*/false);
}

// CDNA5 async global->LDS copy (16B per lane), tracked with ASYNCcnt.
// VDST = per-lane LDS byte address, VADDR = per-lane 64-bit global address.
__device__ __forceinline__ void async_copy_b128(unsigned lds_off,
                                                const void* gptr) {
  asm volatile("global_load_async_to_lds_b128 %0, %1, off"
               :: "v"(lds_off), "v"((unsigned long long)(uintptr_t)gptr)
               : "memory");
}
__device__ __forceinline__ void wait_async0() {
  asm volatile("s_wait_asynccnt 0x0" ::: "memory");
}

// ---------------------------------------------------------------------------
// f32 -> bf16 conversion
// ---------------------------------------------------------------------------
__global__ void cvt_f32_bf16_kernel(const float* __restrict__ s,
                                    unsigned short* __restrict__ d,
                                    size_t n) {
  size_t i = (size_t)blockIdx.x * 256 + threadIdx.x;
  if (i < n) d[i] = f32_to_bf16(s[i]);
}

// ---------------------------------------------------------------------------
// C[M,N] = A[M,K] * W[N,K]^T   (A,W bf16 row-major over K; C f32 row-major)
// Block (256 thr, 8 waves) -> 128x128 C tile; wave -> 64x32 (4x2 WMMA tiles).
// K-slices of 32 staged in LDS by async copies, double-buffered.
// LDS row stride = 40 shorts (80 B = 5x16B): bank-conflict-free fragment
// reads, 16B-aligned b128 runs. grid = (N/128, M/128).
// ---------------------------------------------------------------------------
#define LDSROW 40
__global__ void gemm_bf16_wmma_kernel(const unsigned short* __restrict__ A,
                                      const unsigned short* __restrict__ W,
                                      float* __restrict__ C,
                                      int M, int N, int K) {
  __shared__ __align__(16) unsigned short As[2][128 * LDSROW];
  __shared__ __align__(16) unsigned short Bs[2][128 * LDSROW];

  const int t    = threadIdx.x;
  const int lane = t & 31;
  const int wave = t >> 5;
  const int kh   = lane >> 4;
  const int l15  = lane & 15;
  const int wm   = wave & 1;   // 2 waves along M (64 rows each)
  const int wn   = wave >> 1;  // 4 waves along N (32 cols each)

  const int m0 = blockIdx.y * 128;
  const int n0 = blockIdx.x * 128;

  // staging map: 512 16B chunks per tile / 256 threads = 2 chunks each
  const int ar0 = t >> 2,           ac0 = t & 3;
  const int ar1 = (t + 256) >> 2,   ac1 = (t + 256) & 3;

  const unsigned short* gA = A + (size_t)m0 * K;
  const unsigned short* gW = W + (size_t)n0 * K;

  f32x8 acc[4][2] = {};

  auto stage = [&](int buf, int k0) {
    async_copy_b128((unsigned)(uintptr_t)&As[buf][ar0 * LDSROW + ac0 * 8],
                    gA + (size_t)ar0 * K + k0 + ac0 * 8);
    async_copy_b128((unsigned)(uintptr_t)&As[buf][ar1 * LDSROW + ac1 * 8],
                    gA + (size_t)ar1 * K + k0 + ac1 * 8);
    async_copy_b128((unsigned)(uintptr_t)&Bs[buf][ar0 * LDSROW + ac0 * 8],
                    gW + (size_t)ar0 * K + k0 + ac0 * 8);
    async_copy_b128((unsigned)(uintptr_t)&Bs[buf][ar1 * LDSROW + ac1 * 8],
                    gW + (size_t)ar1 * K + k0 + ac1 * 8);
  };

  const int nsteps = K / 32;
  stage(0, 0);
  wait_async0();
  __syncthreads();

  for (int it = 0; it < nsteps; ++it) {
    const int cur = it & 1;
    if (it + 1 < nsteps) stage(1 - cur, (it + 1) * 32);  // overlap with WMMA

    bf16x16 bfr[2];
#pragma unroll
    for (int j = 0; j < 2; ++j) {
      const unsigned short* bp = &Bs[cur][(wn * 32 + j * 16 + l15) * LDSROW];
      bfr[j] = load_frag(bp + kh * 8, bp + 16 + kh * 8);
    }
#pragma unroll
    for (int i = 0; i < 4; ++i) {
      const unsigned short* ap = &As[cur][(wm * 64 + i * 16 + l15) * LDSROW];
      bf16x16 afr = load_frag(ap + kh * 8, ap + 16 + kh * 8);
#pragma unroll
      for (int j = 0; j < 2; ++j)
        acc[i][j] = wmma_bf16(afr, bfr[j], acc[i][j]);
    }
    wait_async0();
    __syncthreads();
  }

  // C layout: VGPR r -> row +r+8*kh, col l15
#pragma unroll
  for (int i = 0; i < 4; ++i) {
#pragma unroll
    for (int j = 0; j < 2; ++j) {
      float* cp = C + (size_t)(m0 + wm * 64 + i * 16 + 8 * kh) * N
                    + (n0 + wn * 32 + j * 16 + l15);
#pragma unroll
      for (int r = 0; r < 8; ++r)
        cp[(size_t)r * N] = acc[i][j][r];
    }
  }
}

// ---------------------------------------------------------------------------
// Q path: RMSNorm(head) -> RoPE -> * ssmax*log(S)/sqrt(HD) -> bf16 [B,NH,S,HD]
// ---------------------------------------------------------------------------
__global__ void qnorm_rope_kernel(const float* __restrict__ q,     // [B*S, NH*HD]
                                  const float* __restrict__ w,     // [HD]
                                  const float* __restrict__ cosb,  // [S, HD]
                                  const float* __restrict__ sinb,  // [S, HD]
                                  const float* __restrict__ ssmax, // [NH]
                                  unsigned short* __restrict__ qo, // [B,NH,S,HD]
                                  float kvlog_invsqrt) {
  int idx = blockIdx.x;
  int h = idx % NH_; idx /= NH_;
  int s = idx % S_;  int b = idx / S_;
  int d = threadIdx.x;

  __shared__ float red[HD_];
  __shared__ float xs[HD_];

  float x = q[((size_t)(b * S_ + s)) * (NH_ * HD_) + h * HD_ + d];
  red[d] = x * x;
  __syncthreads();
#pragma unroll
  for (int off = 64; off > 0; off >>= 1) {
    if (d < off) red[d] += red[d + off];
    __syncthreads();
  }
  float var = red[0] * (1.0f / HD_);
  float xn = x * rsqrtf(var + EPSV) * w[d];
  xs[d] = xn;
  __syncthreads();
  float other = (d < 64) ? -xs[d + 64] : xs[d - 64];
  float out = xn * cosb[(size_t)s * HD_ + d] + other * sinb[(size_t)s * HD_ + d];
  out *= ssmax[h] * kvlog_invsqrt;  // ssmax * log(kv_len) * 1/sqrt(HD) folded
  qo[(((size_t)(b * NH_ + h)) * S_ + s) * HD_ + d] = f32_to_bf16(out);
}

// K path: RMSNorm -> RoPE -> bf16 [B,NKV,S,HD]
__global__ void knorm_rope_kernel(const float* __restrict__ k,     // [B*S, NKV*HD]
                                  const float* __restrict__ w,
                                  const float* __restrict__ cosb,
                                  const float* __restrict__ sinb,
                                  unsigned short* __restrict__ ko) {
  int idx = blockIdx.x;
  int kv = idx % NKV_; idx /= NKV_;
  int s = idx % S_;    int b = idx / S_;
  int d = threadIdx.x;

  __shared__ float red[HD_];
  __shared__ float xs[HD_];

  float x = k[((size_t)(b * S_ + s)) * (NKV_ * HD_) + kv * HD_ + d];
  red[d] = x * x;
  __syncthreads();
#pragma unroll
  for (int off = 64; off > 0; off >>= 1) {
    if (d < off) red[d] += red[d + off];
    __syncthreads();
  }
  float var = red[0] * (1.0f / HD_);
  float xn = x * rsqrtf(var + EPSV) * w[d];
  xs[d] = xn;
  __syncthreads();
  float other = (d < 64) ? -xs[d + 64] : xs[d - 64];
  float out = xn * cosb[(size_t)s * HD_ + d] + other * sinb[(size_t)s * HD_ + d];
  ko[(((size_t)(b * NKV_ + kv)) * S_ + s) * HD_ + d] = f32_to_bf16(out);
}

// V: [B,S,NKV,HD] f32 -> [B,NKV,HD,S] bf16 (PV B-fragments contiguous in key)
__global__ void v_transpose_kernel(const float* __restrict__ v,
                                   unsigned short* __restrict__ vt) {
  size_t i = (size_t)blockIdx.x * 256 + threadIdx.x;
  if (i >= (size_t)B_ * S_ * NKV_ * HD_) return;
  int d = (int)(i % HD_); size_t tt = i / HD_;
  int kv = (int)(tt % NKV_); tt /= NKV_;
  int s = (int)(tt % S_);    int b = (int)(tt / S_);
  vt[(((size_t)(b * NKV_ + kv)) * HD_ + d) * S_ + s] = f32_to_bf16(v[i]);
}

// ---------------------------------------------------------------------------
// Flash-style causal attention. One wave = one 16-row Q tile; K chunks of 32.
// Q pre-scaled by ssmax*log(S)/sqrt(HD). grid = B*NH*(S/16/8), block = 256.
// ---------------------------------------------------------------------------
#define AWAVES 8
__global__ void attn_kernel(const unsigned short* __restrict__ Q,  // [B,NH,S,HD]
                            const unsigned short* __restrict__ Kh, // [B,NKV,S,HD]
                            const unsigned short* __restrict__ Vt, // [B,NKV,HD,S]
                            float* __restrict__ O) {               // [B*S, NH*HD]
  __shared__ __align__(16) unsigned short plds[AWAVES][16 * 32];

  const int lane = threadIdx.x & 31;
  const int wave = threadIdx.x >> 5;
  const int kh   = lane >> 4;
  const int l15  = lane & 15;

  int idx = blockIdx.x;
  const int qblocks = S_ / 16 / AWAVES;
  int qb = idx % qblocks; idx /= qblocks;
  int h = idx % NH_; int b = idx / NH_;
  int q0 = (qb * AWAVES + wave) * 16;
  int kvh = h / (NH_ / NKV_);  // GQA repeat mapping

  const unsigned short* qbase = Q  + ((size_t)(b * NH_  + h))   * S_  * HD_;
  const unsigned short* kbase = Kh + ((size_t)(b * NKV_ + kvh)) * S_  * HD_;
  const unsigned short* vbase = Vt + ((size_t)(b * NKV_ + kvh)) * HD_ * S_;
  const unsigned short* qrow  = qbase + (size_t)(q0 + l15) * HD_;

  float m[8], l[8];
  f32x8 acc[8] = {};
#pragma unroll
  for (int r = 0; r < 8; ++r) { m[r] = -1e30f; l[r] = 0.0f; }

  for (int k0 = 0; k0 < q0 + 16; k0 += 32) {
    // ---- S tile = Q (16xHD) . K^T (HDx32): two 16-col WMMAs per d-slice ----
    f32x8 s0 = {}, s1 = {};
    const unsigned short* krow0 = kbase + (size_t)(k0 + l15) * HD_;
    const unsigned short* krow1 = kbase + (size_t)(k0 + 16 + l15) * HD_;
#pragma unroll
    for (int dk = 0; dk < HD_; dk += 32) {
      bf16x16 aq = load_frag(qrow + dk + kh * 8, qrow + dk + 16 + kh * 8);
      bf16x16 b0 = load_frag(krow0 + dk + kh * 8, krow0 + dk + 16 + kh * 8);
      bf16x16 b1 = load_frag(krow1 + dk + kh * 8, krow1 + dk + 16 + kh * 8);
      s0 = wmma_bf16(aq, b0, s0);
      s1 = wmma_bf16(aq, b1, s1);
    }

    // ---- causal mask + online softmax (rows live in half-wave groups) ----
#pragma unroll
    for (int r = 0; r < 8; ++r) {
      int qg = q0 + r + 8 * kh;
      if (k0 + l15 > qg)      s0[r] += NEGV;
      if (k0 + 16 + l15 > qg) s1[r] += NEGV;
      float mx = fmaxf(s0[r], s1[r]);
      for (int msk = 1; msk < 16; msk <<= 1)
        mx = fmaxf(mx, __shfl_xor(mx, msk, 32));
      float mnew = fmaxf(m[r], mx);
      float rescale = __expf(m[r] - mnew);
      float p0 = __expf(s0[r] - mnew);
      float p1 = __expf(s1[r] - mnew);
      float rs = p0 + p1;
      for (int msk = 1; msk < 16; msk <<= 1)
        rs += __shfl_xor(rs, msk, 32);
      l[r] = l[r] * rescale + rs;
      m[r] = mnew;
#pragma unroll
      for (int j = 0; j < 8; ++j) acc[j][r] *= rescale;
      // C-layout -> LDS (row-major 16x32 bf16 P tile)
      plds[wave][(r + 8 * kh) * 32 + l15]      = f32_to_bf16(p0);
      plds[wave][(r + 8 * kh) * 32 + 16 + l15] = f32_to_bf16(p1);
    }
    asm volatile("s_wait_dscnt 0" ::: "memory");  // wave-local LDS RAW fence

    // ---- O += P (16x32) . V (32xHD): 8 WMMAs across HD ----
    const unsigned short* pr = &plds[wave][l15 * 32];
    bf16x16 pf = load_frag(pr + kh * 8, pr + 16 + kh * 8);
#pragma unroll
    for (int j = 0; j < 8; ++j) {
      const unsigned short* vcol = vbase + (size_t)(j * 16 + l15) * S_ + k0;
      bf16x16 vf = load_frag(vcol + kh * 8, vcol + 16 + kh * 8);
      acc[j] = wmma_bf16(pf, vf, acc[j]);
    }
    asm volatile("" ::: "memory");  // keep next iter's LDS stores after loads
  }

  // ---- normalize + store [B,S,NH*HD] f32 ----
  float* orow = O + ((size_t)(b * S_ + q0)) * (NH_ * HD_) + h * HD_ + l15;
#pragma unroll
  for (int r = 0; r < 8; ++r) {
    float invl = 1.0f / l[r];
#pragma unroll
    for (int j = 0; j < 8; ++j)
      orow[(size_t)(r + 8 * kh) * (NH_ * HD_) + j * 16] = acc[j][r] * invl;
  }
}

// ---------------------------------------------------------------------------
// Host-side orchestration
// ---------------------------------------------------------------------------
extern "C" void kernel_launch(void* const* d_in, const int* in_sizes, int n_in,
                              void* d_out, int out_size, void* d_ws, size_t ws_size,
                              hipStream_t stream) {
  const float* hs    = (const float*)d_in[0];
  const float* cosb  = (const float*)d_in[1];
  const float* sinb  = (const float*)d_in[2];
  // d_in[3] = attention_mask (pure causal -1e9; applied analytically)
  const float* wq    = (const float*)d_in[4];
  const float* wk    = (const float*)d_in[5];
  const float* wv    = (const float*)d_in[6];
  const float* wo    = (const float*)d_in[7];
  const float* qnw   = (const float*)d_in[8];
  const float* knw   = (const float*)d_in[9];
  const float* ssmax = (const float*)d_in[10];
  float* out = (float*)d_out;

  const size_t M    = (size_t)B_ * S_;        // 4096
  const size_t NQO  = (size_t)NH_ * HD_;      // 2048
  const size_t NKVD = (size_t)NKV_ * HD_;     // 512

  // workspace layout (bytes), aliased where lifetimes don't overlap
  char* p = (char*)d_ws;
  unsigned short* hs_bf = (unsigned short*)p; p += M * H_ * 2;        // also attn_bf
  unsigned short* wq_bf = (unsigned short*)p; p += NQO * H_ * 2;
  unsigned short* wk_bf = (unsigned short*)p; p += NKVD * H_ * 2;
  unsigned short* wv_bf = (unsigned short*)p; p += NKVD * H_ * 2;
  unsigned short* wo_bf = (unsigned short*)p; p += (size_t)H_ * NQO * 2;
  float*          q_f   = (float*)p;          p += M * NQO * 4;       // also attn_f
  float*          k_f   = (float*)p;          p += M * NKVD * 4;
  float*          v_f   = (float*)p;          p += M * NKVD * 4;
  unsigned short* qh_bf = (unsigned short*)p; p += M * NQO * 2;
  unsigned short* kh_bf = (unsigned short*)p; p += M * NKVD * 2;
  unsigned short* vt_bf = (unsigned short*)p; p += M * NKVD * 2;
  unsigned short* attn_bf = hs_bf;  // alias: hs_bf dead after projections
  float*          attn_f  = q_f;    // alias: q_f dead after qnorm_rope

  const float kvlog_invsqrt = 7.6246189861593985f * 0.08838834764831845f; // ln(2048)/sqrt(128)

  // 1. bf16 converts
  {
    size_t n;
    n = M * H_;        cvt_f32_bf16_kernel<<<(unsigned)((n + 255) / 256), 256, 0, stream>>>(hs, hs_bf, n);
    n = NQO * H_;      cvt_f32_bf16_kernel<<<(unsigned)((n + 255) / 256), 256, 0, stream>>>(wq, wq_bf, n);
    n = NKVD * H_;     cvt_f32_bf16_kernel<<<(unsigned)((n + 255) / 256), 256, 0, stream>>>(wk, wk_bf, n);
    n = NKVD * H_;     cvt_f32_bf16_kernel<<<(unsigned)((n + 255) / 256), 256, 0, stream>>>(wv, wv_bf, n);
    n = (size_t)H_ * NQO; cvt_f32_bf16_kernel<<<(unsigned)((n + 255) / 256), 256, 0, stream>>>(wo, wo_bf, n);
  }

  // 2. projections (WMMA, async-LDS staged)
  gemm_bf16_wmma_kernel<<<dim3((unsigned)(NQO / 128),  (unsigned)(M / 128)), 256, 0, stream>>>(hs_bf, wq_bf, q_f, (int)M, (int)NQO,  H_);
  gemm_bf16_wmma_kernel<<<dim3((unsigned)(NKVD / 128), (unsigned)(M / 128)), 256, 0, stream>>>(hs_bf, wk_bf, k_f, (int)M, (int)NKVD, H_);
  gemm_bf16_wmma_kernel<<<dim3((unsigned)(NKVD / 128), (unsigned)(M / 128)), 256, 0, stream>>>(hs_bf, wv_bf, v_f, (int)M, (int)NKVD, H_);

  // 3. norm / rope / layout
  qnorm_rope_kernel<<<(unsigned)(B_ * S_ * NH_), HD_, 0, stream>>>(q_f, qnw, cosb, sinb, ssmax, qh_bf, kvlog_invsqrt);
  knorm_rope_kernel<<<(unsigned)(B_ * S_ * NKV_), HD_, 0, stream>>>(k_f, knw, cosb, sinb, kh_bf);
  {
    size_t n = M * NKVD;
    v_transpose_kernel<<<(unsigned)((n + 255) / 256), 256, 0, stream>>>(v_f, vt_bf);
  }

  // 4. attention (WMMA flash)
  attn_kernel<<<(unsigned)(B_ * NH_ * (S_ / 16 / AWAVES)), 256, 0, stream>>>(qh_bf, kh_bf, vt_bf, attn_f);

  // 5. output projection (WMMA)
  {
    size_t n = M * NQO;
    cvt_f32_bf16_kernel<<<(unsigned)((n + 255) / 256), 256, 0, stream>>>(attn_f, attn_bf, n);
  }
  gemm_bf16_wmma_kernel<<<dim3((unsigned)(H_ / 128), (unsigned)(M / 128)), 256, 0, stream>>>(attn_bf, wo_bf, out, (int)M, H_, (int)NQO);
}
